// InitLayer_85744727097811
// MI455X (gfx1250) — compile-verified, matrix operand order checked by reading, not run on hardware
//
#include <hip/hip_runtime.h>
#include <hip/hip_bf16.h>
#include <math.h>

typedef __attribute__((ext_vector_type(16))) __bf16 v16bf;
typedef __attribute__((ext_vector_type(8)))  float  v8f;

namespace {

constexpr int E_EDGES  = 640000;
constexpr int N_NODES  = 20000;
constexpr int FEAT     = 288;        // 32*1 + 32*3 + 32*5
constexpr int LSTRIDE  = 160;        // LDS row stride in bf16 (K1 padded 136->160)

// workspace byte offsets
constexpr size_t WS_NODESUM = 0;                       // float [20000][288] = 23,040,000 B
constexpr size_t WS_W1T     = 23040000;                // bf16 [128][160]
constexpr size_t WS_W2T     = WS_W1T + 128*160*2;      // bf16 [128][128]
constexpr size_t WS_W3T     = WS_W2T + 128*128*2;      // bf16 [128][128]
constexpr size_t WS_WET     = WS_W3T + 128*128*2;      // bf16 [ 96][128]

// ---- WMMA fragment assembly -------------------------------------------------
// 16-bit A (16x32) / B^T layout: lane l -> row r=l&15, half hi=l>>4.
// VGPR 0..3 hold k = kbase + hi*8 + 0..7 ; VGPR 4..7 hold k = kbase + 16 + hi*8 + 0..7.
// => two contiguous 16-byte loads per fragment.
__device__ __forceinline__ v16bf load_frag(const __bf16* p, int row, int stride,
                                           int kbase, int hi) {
  const __bf16* q = p + row * stride + kbase + hi * 8;
  union { uint4 u[2]; v16bf v; } f;
  f.u[0] = *(const uint4*)(q);
  f.u[1] = *(const uint4*)(q + 16);
  return f.v;
}

__device__ __forceinline__ v8f gemm_tile(const __bf16* wT, int wstride, int n, int hi,
                                         const v16bf* afr, int nk) {
  v8f acc = {0.f, 0.f, 0.f, 0.f, 0.f, 0.f, 0.f, 0.f};
#pragma unroll
  for (int ks = 0; ks < 5; ++ks) {
    if (ks >= nk) break;
    v16bf b = load_frag(wT, n, wstride, ks * 32, hi);
    acc = __builtin_amdgcn_wmma_f32_16x16x32_bf16(false, afr[ks], false, b,
                                                  (short)0, acc, false, false);
  }
  return acc;
}

__device__ __forceinline__ float silu(float x) {
  return x / (1.f + __expf(-x));
}

// ---- prep kernels -----------------------------------------------------------
__global__ void zero_kernel(float* p, int n) {
  int i = blockIdx.x * blockDim.x + threadIdx.x;
  int stride = gridDim.x * blockDim.x;
  for (; i < n; i += stride) p[i] = 0.f;
}

__global__ void convert_weights_kernel(const float* W1, const float* W2,
                                       const float* W3, const float* We,
                                       __bf16* w1t, __bf16* w2t, __bf16* w3t,
                                       __bf16* wet) {
  int tid = blockIdx.x * blockDim.x + threadIdx.x;
  int stride = gridDim.x * blockDim.x;
  // W1T[n][k] (n<128, k<160), W1 is [136][128]; pad k>=136 with 0
  for (int i = tid; i < 128 * 160; i += stride) {
    int n = i / 160, k = i % 160;
    w1t[i] = (__bf16)((k < 136) ? W1[k * 128 + n] : 0.f);
  }
  for (int i = tid; i < 128 * 128; i += stride) {
    int n = i >> 7, k = i & 127;
    w2t[i] = (__bf16)W2[k * 128 + n];
    w3t[i] = (__bf16)W3[k * 128 + n];
  }
  // WenvT[n][k] (n<96, k<128), W_env is [128][96]
  for (int i = tid; i < 96 * 128; i += stride) {
    int n = i >> 7, k = i & 127;
    wet[i] = (__bf16)We[k * 96 + n];
  }
}

// ---- fused edge pipeline ----------------------------------------------------
// 128 edges / block; wave w owns the 16-edge strip [blockIdx*128 + 16w, +16).
__global__ __launch_bounds__(256) void edge_mlp_kernel(
    const int* __restrict__ edge_index, const float* __restrict__ edge_sh,
    const float* __restrict__ edge_length, const float* __restrict__ one_hot,
    const float* __restrict__ bessel_w,
    const __bf16* __restrict__ w1t, const __bf16* __restrict__ w2t,
    const __bf16* __restrict__ w3t, const __bf16* __restrict__ wet,
    float* __restrict__ out_raw, float* __restrict__ out_edge,
    float* __restrict__ node_sum) {
  __shared__ alignas(16) __bf16 sbuf[8 * 2 * 16 * LSTRIDE];  // 81,920 B

  const int wave = threadIdx.x >> 5;
  const int lane = threadIdx.x & 31;
  const int r16  = lane & 15;
  const int hi   = lane >> 4;
  const int edge0 = blockIdx.x * 128 + wave * 16;

  __bf16* buf0 = &sbuf[wave * 2 * 16 * LSTRIDE];
  __bf16* buf1 = buf0 + 16 * LSTRIDE;

  // ---- stage 0: build h0 = [one_hot(128) | bessel(8) | pad(24)] as bf16 ----
  for (int t = lane; t < 16 * LSTRIDE; t += 32) {
    int r = t / LSTRIDE, c = t - r * LSTRIDE;
    int e = edge0 + r;
    float v;
    if (c < 128) {
      v = one_hot[e * 128 + c];
    } else if (c < 136) {
      float x = edge_length[e];
      v = 0.4f * __sinf(bessel_w[c - 128] * x * 0.2f) / x;  // 2/R * sin(w x/R)/x
    } else {
      v = 0.f;
    }
    buf0[t] = (__bf16)v;
  }
  asm volatile("s_wait_dscnt 0x0" ::: "memory");

  // ---- stage 1: h1 = silu(h0 @ W1)   (K=160, N=128) ----
  {
    v16bf afr[5];
#pragma unroll
    for (int ks = 0; ks < 5; ++ks) afr[ks] = load_frag(buf0, r16, LSTRIDE, ks * 32, hi);
#pragma unroll
    for (int nt = 0; nt < 8; ++nt) {
      int n = r16 + nt * 16;
      v8f acc = gemm_tile(w1t, 160, n, hi, afr, 5);
#pragma unroll
      for (int i = 0; i < 8; ++i)
        buf1[(i + 8 * hi) * LSTRIDE + n] = (__bf16)silu(acc[i]);
    }
  }
  asm volatile("s_wait_dscnt 0x0" ::: "memory");

  // ---- stage 2: h2 = silu(h1 @ W2)   (K=128, N=128) ----
  {
    v16bf afr[5];
#pragma unroll
    for (int ks = 0; ks < 4; ++ks) afr[ks] = load_frag(buf1, r16, LSTRIDE, ks * 32, hi);
#pragma unroll
    for (int nt = 0; nt < 8; ++nt) {
      int n = r16 + nt * 16;
      v8f acc = gemm_tile(w2t, 128, n, hi, afr, 4);
#pragma unroll
      for (int i = 0; i < 8; ++i)
        buf0[(i + 8 * hi) * LSTRIDE + n] = (__bf16)silu(acc[i]);
    }
  }
  asm volatile("s_wait_dscnt 0x0" ::: "memory");

  // ---- stage 3: raw = h2 @ W3  (K=128, N=128); f32 -> out_raw, bf16 -> buf1 ----
  {
    v16bf afr[5];
#pragma unroll
    for (int ks = 0; ks < 4; ++ks) afr[ks] = load_frag(buf0, r16, LSTRIDE, ks * 32, hi);
#pragma unroll
    for (int nt = 0; nt < 8; ++nt) {
      int n = r16 + nt * 16;
      v8f acc = gemm_tile(w3t, 128, n, hi, afr, 4);
#pragma unroll
      for (int i = 0; i < 8; ++i) {
        int m = i + 8 * hi;
        out_raw[(edge0 + m) * 128 + n] = acc[i];
        buf1[m * LSTRIDE + n] = (__bf16)acc[i];
      }
    }
  }
  asm volatile("s_wait_dscnt 0x0" ::: "memory");

  // ---- stage 4: weights_e = raw @ W_env  (K=128, N=96) ----
  float* fwe = (float*)buf0;  // [16][96] f32 = 6144 B (spans buf0 + head of buf1)
  {
    v16bf afr[5];
#pragma unroll
    for (int ks = 0; ks < 4; ++ks) afr[ks] = load_frag(buf1, r16, LSTRIDE, ks * 32, hi);
    v8f acc6[6];
#pragma unroll
    for (int nt = 0; nt < 6; ++nt)
      acc6[nt] = gemm_tile(wet, 128, r16 + nt * 16, hi, afr, 4);
    // all A-fragments consumed (acc6 depends on them) -> safe to overwrite LDS
#pragma unroll
    for (int nt = 0; nt < 6; ++nt)
#pragma unroll
      for (int i = 0; i < 8; ++i)
        fwe[(i + 8 * hi) * 96 + r16 + nt * 16] = acc6[nt][i];
  }
  asm volatile("s_wait_dscnt 0x0" ::: "memory");

  // ---- stage 5: edge_features = w (x) sh ; stream out + scatter-add to nodes ----
  for (int t = lane; t < 16 * FEAT; t += 32) {
    int r = t / FEAT, c = t - r * FEAT;
    int e = edge0 + r;
    int l, k, m, off;
    if (c < 32)       { l = 0; k = c;          m = 0;          off = 0; }
    else if (c < 128) { int q = c - 32;  l = 1; k = q / 3; m = q - 3 * k; off = 1; }
    else              { int q = c - 128; l = 2; k = q / 5; m = q - 5 * k; off = 4; }
    float val = fwe[r * 96 + l * 32 + k] * edge_sh[e * 9 + off + m];
    out_edge[e * FEAT + c] = val;
    int node = edge_index[e];  // row 0 of [2,E]
    __hip_atomic_fetch_add(&node_sum[node * FEAT + c], val,
                           __ATOMIC_RELAXED, __HIP_MEMORY_SCOPE_AGENT);
  }
}

// ---- node layer-norm / equivariant norm (one wave per node) -----------------
__global__ __launch_bounds__(256) void node_norm_kernel(
    const float* __restrict__ node_sum, const float* __restrict__ g0,
    const float* __restrict__ b0, const float* __restrict__ gl,
    float* __restrict__ out_node) {
  const int wave = threadIdx.x >> 5;
  const int lane = threadIdx.x & 31;
  const int v = blockIdx.x * 8 + wave;
  if (v >= N_NODES) return;

  const float scale = 0.17677669529663687f;  // 1/sqrt(32)
  const float* row = node_sum + (size_t)v * FEAT;

  // scalar part: LayerNorm over 32 channels (one per lane)
  float s = row[lane] * scale;
  float mu = s;
#pragma unroll
  for (int m = 16; m; m >>= 1) mu += __shfl_xor(mu, m, 32);
  mu *= (1.f / 32.f);
  float d = s - mu;
  float var = d * d;
#pragma unroll
  for (int m = 16; m; m >>= 1) var += __shfl_xor(var, m, 32);
  var *= (1.f / 32.f);
  float inv0 = rsqrtf(var + 1e-8f);
  out_node[(size_t)v * FEAT + lane] = d * inv0 * g0[lane] + b0[lane];

  // vector part: joint equivariant norm over v1 (96) + v2 (160)
  float vals[8];
  float acc = 0.f;
#pragma unroll
  for (int j = 0; j < 8; ++j) {
    int t = lane + 32 * j;
    float x = row[32 + t] * scale;
    vals[j] = x;
    acc += x * x * ((t < 96) ? (1.f / 6.f) : (1.f / 10.f));
  }
#pragma unroll
  for (int m = 16; m; m >>= 1) acc += __shfl_xor(acc, m, 32);
  float inv = rsqrtf(acc * (1.f / 32.f) + 1e-8f);
#pragma unroll
  for (int j = 0; j < 8; ++j) {
    int t = lane + 32 * j;
    int gidx = (t < 96) ? (t / 3) : (32 + (t - 96) / 5);
    out_node[(size_t)v * FEAT + 32 + t] = vals[j] * inv * gl[gidx];
  }
}

}  // namespace

extern "C" void kernel_launch(void* const* d_in, const int* in_sizes, int n_in,
                              void* d_out, int out_size, void* d_ws, size_t ws_size,
                              hipStream_t stream) {
  const int*   edge_index  = (const int*)d_in[0];
  // d_in[1] atom_type: unused by the forward pass
  const float* edge_sh     = (const float*)d_in[2];
  const float* edge_length = (const float*)d_in[3];
  const float* one_hot     = (const float*)d_in[4];
  const float* bessel_w    = (const float*)d_in[5];
  const float* W1          = (const float*)d_in[6];
  const float* W2          = (const float*)d_in[7];
  const float* W3          = (const float*)d_in[8];
  const float* Wenv        = (const float*)d_in[9];
  const float* ln_g0       = (const float*)d_in[10];
  const float* ln_b0       = (const float*)d_in[11];
  const float* ln_gl       = (const float*)d_in[12];

  float* out      = (float*)d_out;
  float* out_raw  = out;                            // [E,128]
  float* out_node = out + (size_t)E_EDGES * 128;    // [N,288]
  float* out_edge = out_node + (size_t)N_NODES * FEAT;  // [E,288]

  char*   ws       = (char*)d_ws;
  float*  node_sum = (float*)(ws + WS_NODESUM);
  __bf16* w1t      = (__bf16*)(ws + WS_W1T);
  __bf16* w2t      = (__bf16*)(ws + WS_W2T);
  __bf16* w3t      = (__bf16*)(ws + WS_W3T);
  __bf16* wet      = (__bf16*)(ws + WS_WET);

  zero_kernel<<<2048, 256, 0, stream>>>(node_sum, N_NODES * FEAT);
  convert_weights_kernel<<<64, 256, 0, stream>>>(W1, W2, W3, Wenv, w1t, w2t, w3t, wet);
  edge_mlp_kernel<<<E_EDGES / 128, 256, 0, stream>>>(
      edge_index, edge_sh, edge_length, one_hot, bessel_w,
      w1t, w2t, w3t, wet, out_raw, out_edge, node_sum);
  node_norm_kernel<<<(N_NODES + 7) / 8, 256, 0, stream>>>(
      node_sum, ln_g0, ln_b0, ln_gl, out_node);
}
